// PhysicalHarmonicGCN_20796231647470
// MI455X (gfx1250) — compile-verified, matrix operand order checked by reading, not run on hardware
//
#include <hip/hip_runtime.h>
#include <hip/hip_bf16.h>

typedef __bf16 bf16_t;
typedef __attribute__((ext_vector_type(16))) __bf16 v16bf;
typedef __attribute__((ext_vector_type(8)))  __bf16 bf16x8;
typedef __attribute__((ext_vector_type(8)))  float  v8f;
typedef __attribute__((ext_vector_type(4)))  float  f32x4;

#define BATCH 256
#define NNODE 180
#define NPAD  192
#define CDIM  768
#define RDIM  192

// Load one 16x32 bf16 WMMA fragment row for this lane.
// rowptr -> element [row][k0] of a row-major matrix (16B-aligned rows),
// g = 8*(lane>>4). Elements 0..7 = K[g..g+7], 8..15 = K[g+16..g+23],
// matching the CDNA5 16-bit A/B VGPR layout (B read from a transposed layout).
__device__ __forceinline__ v16bf frag_ld(const bf16_t* rowptr, int g) {
  bf16x8 a = *(const bf16x8*)(rowptr + g);
  bf16x8 b = *(const bf16x8*)(rowptr + g + 16);
  v16bf r;
#pragma unroll
  for (int i = 0; i < 8; ++i) { r[i] = a[i]; r[i + 8] = b[i]; }
  return r;
}

__device__ __forceinline__ v8f wmma_bf16(v16bf a, v16bf b, v8f c) {
  return __builtin_amdgcn_wmma_f32_16x16x32_bf16(false, a, false, b, (short)0, c,
                                                 false, false);
}

// ---------------------------------------------------------------------------
// K1: GEMM over padded rows (B*192) x Nc: out = bf16( x @ W + bias ).
// TRANS=false: out row-major [B*192][Nc] (Q, K).
// TRANS=true : out transposed [B][Nc][NPAD] (XG = x@Wg), enabling b128 stores
//              and LDS-free B fragments in k_aggr. Bias ignored when TRANS.
// Rows n>=180 are zero-filled. Grid = (Nc/64, B*192/128): N-tiles fastest so
// the x row-tile stays resident in L2 across its N sweep.
// ---------------------------------------------------------------------------
template <bool TRANS>
__global__ __launch_bounds__(256)
void k_proj(const float* __restrict__ x, const float* __restrict__ W,
            const float* __restrict__ bias, bf16_t* __restrict__ out, int Nc) {
  __shared__ bf16_t xs[128][40];  // A tile (bf16), padded stride
  __shared__ bf16_t wt[64][40];   // B tile transposed: wt[n][k]
  const int tid    = threadIdx.x;
  const int n_base = blockIdx.x * 64;
  const int m_base = blockIdx.y * 128;
  const int wave = tid >> 5, lane = tid & 31;
  const int wm = wave >> 1, wn = wave & 1;         // 4 x 2 wave grid
  const int lr = lane & 15, lg = (lane >> 4) * 8;  // frag row / k-group

  // Hoisted staging descriptors (k-invariant).
  const float* srcx[4];
  int xrow[4], xc4[4];
  bool vx[4];
#pragma unroll
  for (int j = 0; j < 4; ++j) {
    int i = tid + j * 256;           // 1024 float4 groups: 128 rows x 8 groups
    int row = i >> 3, c4 = (i & 7) * 4;
    int gr = m_base + row;
    int b = gr / NPAD, n = gr % NPAD;
    srcx[j] = x + ((size_t)b * NNODE + (n < NNODE ? n : 0)) * CDIM + c4;
    vx[j] = (n < NNODE);
    xrow[j] = row;
    xc4[j] = c4;
  }
  const float* srcw[2];
  int wkk[2], wc4[2];
#pragma unroll
  for (int j = 0; j < 2; ++j) {
    int i = tid + j * 256;           // 512 float4 groups: 32 k x 16 groups
    int k = i >> 4, c4 = (i & 15) * 4;
    srcw[j] = W + (size_t)k * Nc + n_base + c4;
    wkk[j] = k;
    wc4[j] = c4;
  }

  v8f acc[2][2] = {};
  for (int k0 = 0; k0 < CDIM; k0 += 32) {
#pragma unroll
    for (int j = 0; j < 4; ++j) {
      f32x4 v = *(const f32x4*)(srcx[j] + k0);
      if (k0 + 32 < CDIM) __builtin_prefetch(srcx[j] + k0 + 32, 0, 0);
#pragma unroll
      for (int q = 0; q < 4; ++q)
        xs[xrow[j]][xc4[j] + q] = (bf16_t)(vx[j] ? v[q] : 0.0f);
    }
#pragma unroll
    for (int j = 0; j < 2; ++j) {
      f32x4 v = *(const f32x4*)(srcw[j] + (size_t)k0 * Nc);
#pragma unroll
      for (int q = 0; q < 4; ++q) wt[wc4[j] + q][wkk[j]] = (bf16_t)v[q];
    }
    __syncthreads();
    v16bf af[2], bfr[2];
#pragma unroll
    for (int t = 0; t < 2; ++t) {
      af[t]  = frag_ld(&xs[wm * 32 + t * 16 + lr][0], lg);
      bfr[t] = frag_ld(&wt[wn * 32 + t * 16 + lr][0], lg);
    }
#pragma unroll
    for (int tm = 0; tm < 2; ++tm)
#pragma unroll
      for (int tn = 0; tn < 2; ++tn)
        acc[tm][tn] = wmma_bf16(af[tm], bfr[tn], acc[tm][tn]);
    __syncthreads();
  }

#pragma unroll
  for (int tm = 0; tm < 2; ++tm)
#pragma unroll
    for (int tn = 0; tn < 2; ++tn) {
      int coln = n_base + wn * 32 + tn * 16 + lr;
      if (TRANS) {
        // 8 accumulator rows are contiguous in the transposed layout.
        int grow0 = m_base + wm * 32 + tm * 16 + lg;  // row for r=0 (mult of 8)
        int b = grow0 / NPAD, nl = grow0 % NPAD;
        bf16x8 pk;
#pragma unroll
        for (int r = 0; r < 8; ++r) pk[r] = (bf16_t)acc[tm][tn][r];
        *(bf16x8*)(out + ((size_t)b * Nc + coln) * NPAD + nl) = pk;
      } else {
        float bv = bias ? bias[coln] : 0.0f;
#pragma unroll
        for (int r = 0; r < 8; ++r) {
          int grow = m_base + wm * 32 + tm * 16 + r + lg;
          out[(size_t)grow * Nc + coln] = (bf16_t)(acc[tm][tn][r] + bv);
        }
      }
    }
}

// ---------------------------------------------------------------------------
// K2: per (batch, 16-row strip): scores = Q K^T / sqrt(R), softmax over valid
// cols, blend with A_phys, store bf16 A_comb[B][192][192]. One wave per strip,
// fragments loaded directly from global (rows 384B-aligned).
// ---------------------------------------------------------------------------
__global__ __launch_bounds__(256)
void k_attn(const bf16_t* __restrict__ Qb, const bf16_t* __restrict__ Kb,
            const float* __restrict__ A_phys, const float* __restrict__ alpha_p,
            bf16_t* __restrict__ Ac) {
  const int tid = threadIdx.x, lane = tid & 31;
  const int wid = blockIdx.x * 8 + (tid >> 5);
  const int b = wid / 12, strip = wid % 12;
  const int lr = lane & 15, lg = (lane >> 4) * 8;
  const bf16_t* Q = Qb + (size_t)b * NPAD * RDIM;
  const bf16_t* K = Kb + (size_t)b * NPAD * RDIM;

  v8f acc[12] = {};
  const int arow = strip * 16 + lr;
#pragma unroll
  for (int k0 = 0; k0 < RDIM; k0 += 32) {
    v16bf af = frag_ld(Q + (size_t)arow * RDIM + k0, lg);
#pragma unroll
    for (int t = 0; t < 12; ++t) {
      v16bf bfr = frag_ld(K + (size_t)(t * 16 + lr) * RDIM + k0, lg);
      acc[t] = wmma_bf16(af, bfr, acc[t]);
    }
  }
  const float scale = 0.07216878364870323f;  // 1/sqrt(192)
  float a0 = alpha_p[0];
  float sig = 1.0f / (1.0f + __expf(-a0));
#pragma unroll
  for (int t = 0; t < 12; ++t) {
    int col = t * 16 + lr;
#pragma unroll
    for (int r = 0; r < 8; ++r) {
      float v = acc[t][r] * scale;
      acc[t][r] = (col < NNODE) ? v : -1e30f;
    }
  }
  // row softmax: row = strip*16 + r + lg; cross-lane over 16-lane halves
#pragma unroll
  for (int r = 0; r < 8; ++r) {
    float m = -1e30f;
#pragma unroll
    for (int t = 0; t < 12; ++t) m = fmaxf(m, acc[t][r]);
    for (int s = 1; s < 16; s <<= 1) m = fmaxf(m, __shfl_xor(m, s, 32));
    float sum = 0.0f;
#pragma unroll
    for (int t = 0; t < 12; ++t) {
      float e = __expf(acc[t][r] - m);
      acc[t][r] = e;
      sum += e;
    }
    for (int s = 1; s < 16; s <<= 1) sum += __shfl_xor(sum, s, 32);
    float inv = 1.0f / sum;
#pragma unroll
    for (int t = 0; t < 12; ++t) acc[t][r] *= inv;
  }
  bf16_t* outp = Ac + (size_t)b * NPAD * NPAD;
#pragma unroll
  for (int r = 0; r < 8; ++r) {
    int row = strip * 16 + r + lg;
    bool vrow = (row < NNODE);
#pragma unroll
    for (int t = 0; t < 12; ++t) {
      int col = t * 16 + lr;
      float ap = (vrow && col < NNODE) ? A_phys[row * NNODE + col] : 0.0f;
      float res =
          (vrow && col < NNODE) ? (sig * ap + (1.0f - sig) * acc[t][r]) : 0.0f;
      outp[(size_t)row * NPAD + col] = (bf16_t)res;
    }
  }
}

// ---------------------------------------------------------------------------
// K3: z = A_comb @ XG with XG stored transposed [B][C][NPAD]: both A and B
// fragments load straight from global (no LDS, no barriers). Fused epilogue
// h = x + relu(z + bg) -> d_out (f32). Grid = (3 mtiles * 6 ctiles, B):
// tile-blocks of one batch are adjacent so A (72KB) + XG^T (288KB) hit L2.
// ---------------------------------------------------------------------------
__global__ __launch_bounds__(256)
void k_aggr(const bf16_t* __restrict__ Ac, const bf16_t* __restrict__ Gt,
            const float* __restrict__ x, const float* __restrict__ bg,
            float* __restrict__ out) {
  const int b = blockIdx.y;
  const int mt = blockIdx.x / 6, ct = blockIdx.x % 6;
  const int m0 = mt * 64, c0 = ct * 128;
  const int tid = threadIdx.x, wave = tid >> 5, lane = tid & 31;
  const int wm = wave >> 2, wn = wave & 3;  // 2 x 4 waves
  const int lr = lane & 15, lg = (lane >> 4) * 8;
  const bf16_t* A = Ac + (size_t)b * NPAD * NPAD;
  const bf16_t* G = Gt + (size_t)b * CDIM * NPAD;

  v8f acc[2][2] = {};
#pragma unroll
  for (int k0 = 0; k0 < NPAD; k0 += 32) {
    v16bf af[2];
    af[0] = frag_ld(A + (size_t)(m0 + wm * 32 + lr) * NPAD + k0, lg);
    af[1] = frag_ld(A + (size_t)(m0 + wm * 32 + 16 + lr) * NPAD + k0, lg);
#pragma unroll
    for (int tn = 0; tn < 2; ++tn) {
      v16bf bfr =
          frag_ld(G + (size_t)(c0 + wn * 32 + tn * 16 + lr) * NPAD + k0, lg);
      acc[0][tn] = wmma_bf16(af[0], bfr, acc[0][tn]);
      acc[1][tn] = wmma_bf16(af[1], bfr, acc[1][tn]);
    }
  }
#pragma unroll
  for (int tm = 0; tm < 2; ++tm)
#pragma unroll
    for (int tn = 0; tn < 2; ++tn) {
      int col = c0 + wn * 32 + tn * 16 + lr;
      float bgv = bg[col];
#pragma unroll
      for (int r = 0; r < 8; ++r) {
        int n = m0 + wm * 32 + tm * 16 + r + lg;
        if (n < NNODE) {
          size_t idx = ((size_t)b * NNODE + n) * CDIM + col;
          out[idx] = x[idx] + fmaxf(acc[tm][tn][r] + bgv, 0.0f);
        }
      }
    }
}

// ---------------------------------------------------------------------------
// K4: in-place LayerNorm over C=768 per row; one wave per row.
// ---------------------------------------------------------------------------
__global__ __launch_bounds__(256)
void k_ln(float* __restrict__ out, const float* __restrict__ gamma,
          const float* __restrict__ beta) {
  int row = blockIdx.x * 8 + (threadIdx.x >> 5);
  int lane = threadIdx.x & 31;
  if (row >= BATCH * NNODE) return;
  float* p = out + (size_t)row * CDIM;
  float v[24];
  float s = 0.0f;
#pragma unroll
  for (int i = 0; i < 24; ++i) {
    v[i] = p[lane + 32 * i];
    s += v[i];
  }
  for (int m = 1; m < 32; m <<= 1) s += __shfl_xor(s, m, 32);
  float mu = s * (1.0f / CDIM);
  float q = 0.0f;
#pragma unroll
  for (int i = 0; i < 24; ++i) {
    float d = v[i] - mu;
    q += d * d;
  }
  for (int m = 1; m < 32; m <<= 1) q += __shfl_xor(q, m, 32);
  float rs = rsqrtf(q * (1.0f / CDIM) + 1e-5f);
#pragma unroll
  for (int i = 0; i < 24; ++i) {
    int c = lane + 32 * i;
    p[c] = (v[i] - mu) * rs * gamma[c] + beta[c];
  }
}

extern "C" void kernel_launch(void* const* d_in, const int* in_sizes, int n_in,
                              void* d_out, int out_size, void* d_ws,
                              size_t ws_size, hipStream_t stream) {
  (void)in_sizes; (void)n_in; (void)out_size; (void)ws_size;
  const float* x      = (const float*)d_in[0];
  const float* Wq     = (const float*)d_in[1];
  const float* bq     = (const float*)d_in[2];
  const float* Wk     = (const float*)d_in[3];
  const float* bk     = (const float*)d_in[4];
  const float* Wg     = (const float*)d_in[5];
  const float* bg     = (const float*)d_in[6];
  const float* gamma  = (const float*)d_in[7];
  const float* beta   = (const float*)d_in[8];
  const float* alpha  = (const float*)d_in[9];
  const float* A_phys = (const float*)d_in[10];
  float* out = (float*)d_out;

  bf16_t* Qbf = (bf16_t*)d_ws;                      // [B][192][192]
  bf16_t* Kbf = Qbf + (size_t)BATCH * NPAD * RDIM;  // [B][192][192]
  bf16_t* Ac  = Kbf + (size_t)BATCH * NPAD * RDIM;  // [B][192][192]
  bf16_t* XGt = Ac  + (size_t)BATCH * NPAD * NPAD;  // [B][768][192] transposed

  dim3 blk(256);
  const int mtiles = (BATCH * NPAD) / 128;  // 384
  k_proj<false><<<dim3(RDIM / 64, mtiles), blk, 0, stream>>>(x, Wq, bq, Qbf, RDIM);
  k_proj<false><<<dim3(RDIM / 64, mtiles), blk, 0, stream>>>(x, Wk, bk, Kbf, RDIM);
  k_proj<true ><<<dim3(CDIM / 64, mtiles), blk, 0, stream>>>(x, Wg, nullptr, XGt, CDIM);
  k_attn<<<dim3((BATCH * 12) / 8), blk, 0, stream>>>(Qbf, Kbf, A_phys, alpha, Ac);
  k_aggr<<<dim3(18, BATCH), blk, 0, stream>>>(Ac, XGt, x, bg, out);
  k_ln<<<dim3((BATCH * NNODE) / 8), blk, 0, stream>>>(out, gamma, beta);
}
